// PatchEmbedding3D_14955076125179
// MI455X (gfx1250) — matrix operand order repaired and checked by reading, not compiled, not run on hardware
//
#include <hip/hip_runtime.h>
#include <hip/hip_bf16.h>
#include <math.h>

typedef __attribute__((ext_vector_type(16))) _Float16 v16h;
typedef __attribute__((ext_vector_type(8)))  _Float16 v8h;
typedef __attribute__((ext_vector_type(8)))  float    v8f;

#define NUM_POINTS 16384
#define PATCH_SIZE 32
#define EMBED_DIM  256
#define HIDDEN     128
#define N_PATCHES  512
#define BATCH      32
#define LN_EPS     1e-5f

// LDS strides (padded to dodge bank conflicts)
#define H1_ST 136   // halves per row of h1 (128 + 8 pad), 17*8 -> keeps 16B alignment
#define H2_ST 257   // floats per row of h2 (256 + 1 pad)

// ---------------------------------------------------------------------------
// Pre-kernel: repack w2 (128x256 f32, row-major [K][N]) into f16 laid out in
// WMMA-B fragment order:
//   flat idx = (((kc*16 + nt)*32) + lane)*16 + h
//   maps to   K = kc*32 + 8*(lane>>4) + (h&7) + 16*(h>>3) ,  N = nt*16 + (lane&15)
// so the main kernel fetches one contiguous 32-byte v16h per lane.
// ---------------------------------------------------------------------------
__global__ __launch_bounds__(256) void repack_w2_kernel(const float* __restrict__ w2,
                                                        _Float16* __restrict__ w2h) {
    int idx  = blockIdx.x * 256 + threadIdx.x;     // 0 .. 32767
    int h    = idx & 15;
    int lane = (idx >> 4) & 31;
    int tile = idx >> 9;                           // 0 .. 63
    int kc   = tile >> 4;
    int nt   = tile & 15;
    int K    = kc * 32 + 8 * (lane >> 4) + (h & 7) + 16 * (h >> 3);
    int N    = nt * 16 + (lane & 15);
    w2h[idx] = (_Float16)w2[K * 256 + N];
}

__device__ inline v16h cat8(v8h lo, v8h hi) {
    return __builtin_shufflevector(lo, hi, 0, 1, 2, 3, 4, 5, 6, 7,
                                           8, 9, 10, 11, 12, 13, 14, 15);
}

// ---------------------------------------------------------------------------
// Main fused kernel: one workgroup per (batch, patch).
// ---------------------------------------------------------------------------
__global__ __launch_bounds__(256) void patch_embed_kernel(
    const float* __restrict__ x,    // (B, 3, NUM_POINTS)
    const float* __restrict__ w1,   // (3, HIDDEN)
    const float* __restrict__ b1,   // (HIDDEN)
    const float* __restrict__ g1,
    const float* __restrict__ be1,
    const _Float16* __restrict__ w2h, // repacked (see above)
    const float* __restrict__ b2,   // (EMBED_DIM)
    const float* __restrict__ g2,
    const float* __restrict__ be2,
    const float* __restrict__ gn,
    const float* __restrict__ bn,
    float* __restrict__ out)        // (B, N_PATCHES, EMBED_DIM)
{
    __shared__ _Float16 h1[PATCH_SIZE * H1_ST];   // post LN+ReLU activations (f16)
    __shared__ float    h2[PATCH_SIZE * H2_ST];   // GEMM2 output (f32)
    __shared__ float    red[16];                  // cross-wave reduction scratch

    const int tid   = threadIdx.x;
    const int lane  = tid & 31;
    const int wave  = tid >> 5;
    const int blk   = blockIdx.x;                 // b*512 + patch
    const int b     = blk >> 9;
    const int patch = blk & (N_PATCHES - 1);
    const int p0    = patch * PATCH_SIZE;

    // ---------------- Stage A: GEMM1 (K=3) + LayerNorm(128) + ReLU -> LDS f16
    {
        const int row = tid >> 3;                 // 0..31 : point within patch
        const int j0  = (tid & 7) * 16;           // 16 of the 128 hidden cols
        const int pt  = p0 + row;

        const float px = x[(b * 3 + 0) * NUM_POINTS + pt];
        const float py = x[(b * 3 + 1) * NUM_POINTS + pt];
        const float pz = x[(b * 3 + 2) * NUM_POINTS + pt];

        float v[16];
        float sum = 0.f, sq = 0.f;
        #pragma unroll
        for (int i = 0; i < 16; ++i) {
            const int j = j0 + i;
            float t = fmaf(px, w1[j],
                      fmaf(py, w1[HIDDEN + j],
                      fmaf(pz, w1[2 * HIDDEN + j], b1[j])));
            v[i] = t;
            sum += t;
            sq  += t * t;
        }
        // reduce across the 8 threads covering this row (contiguous lanes)
        #pragma unroll
        for (int m = 1; m < 8; m <<= 1) {
            sum += __shfl_xor(sum, m, 8);
            sq  += __shfl_xor(sq,  m, 8);
        }
        const float mean = sum * (1.f / HIDDEN);
        const float var  = sq * (1.f / HIDDEN) - mean * mean;
        const float rs   = rsqrtf(var + LN_EPS);
        #pragma unroll
        for (int i = 0; i < 16; ++i) {
            const int j = j0 + i;
            float t = (v[i] - mean) * rs * g1[j] + be1[j];
            t = fmaxf(t, 0.f);
            h1[row * H1_ST + j] = (_Float16)t;
        }
    }
    __syncthreads();

    // ---------------- Stage B: GEMM2 via v_wmma_f32_16x16x32_f16
    // wave handles N-tiles {2w, 2w+1}, both M-tiles (rows 0-15 and 16-31).
    // A fragments are invariant across the N-tile loop -> load once (8 x v16h).
    {
        const int lh = lane >> 4;                 // 0/1 half-wave
        const int lm = lane & 15;

        // A fragment per ISA layout: halves 0-7 -> K = kc*32 + 8*lh + 0..7,
        //                            halves 8-15 -> +16
        v16h A0[4], A1[4];
        #pragma unroll
        for (int kc = 0; kc < 4; ++kc) {
            const int ka = kc * 32 + 8 * lh;
            const v8h* a0p = (const v8h*)&h1[lm * H1_ST + ka];
            const v8h* a1p = (const v8h*)&h1[(lm + 16) * H1_ST + ka];
            A0[kc] = cat8(a0p[0], a0p[2]);        // +16 halves == +2 v8h
            A1[kc] = cat8(a1p[0], a1p[2]);
        }

        #pragma unroll
        for (int t = 0; t < 2; ++t) {
            const int nt = 2 * wave + t;
            v8f acc0 = {};
            v8f acc1 = {};
            #pragma unroll
            for (int kc = 0; kc < 4; ++kc) {
                // B fragment: one contiguous 32B chunk per lane (pre-repacked)
                v16h bf = *(const v16h*)&w2h[(((kc * 16 + nt) * 32) + lane) * 16];
                acc0 = __builtin_amdgcn_wmma_f32_16x16x32_f16(
                           false, A0[kc], false, bf, (short)0, acc0, false, false);
                acc1 = __builtin_amdgcn_wmma_f32_16x16x32_f16(
                           false, A1[kc], false, bf, (short)0, acc1, false, false);
            }
            // spill C tiles (+ bias) to LDS: VGPR r -> row r + 8*lh (tile0),
            // +16 (tile1); column = nt*16 + lm
            const int   col  = nt * 16 + lm;
            const float bias = b2[col];
            #pragma unroll
            for (int r = 0; r < 8; ++r) {
                h2[(r + 8 * lh) * H2_ST + col]      = acc0[r] + bias;
                h2[(r + 8 * lh + 16) * H2_ST + col] = acc1[r] + bias;
            }
        }
    }
    __syncthreads();

    // ---------------- Stage C: LayerNorm over 256 per row, in place
    {
        const int row = tid >> 3;
        const int c0  = (tid & 7) * 32;
        float sum = 0.f, sq = 0.f;
        #pragma unroll
        for (int i = 0; i < 32; ++i) {
            float t = h2[row * H2_ST + c0 + i];
            sum += t;
            sq  += t * t;
        }
        #pragma unroll
        for (int m = 1; m < 8; m <<= 1) {
            sum += __shfl_xor(sum, m, 8);
            sq  += __shfl_xor(sq,  m, 8);
        }
        const float mean = sum * (1.f / EMBED_DIM);
        const float var  = sq * (1.f / EMBED_DIM) - mean * mean;
        const float rs   = rsqrtf(var + LN_EPS);
        #pragma unroll
        for (int i = 0; i < 32; ++i) {
            const int c = c0 + i;
            float t = h2[row * H2_ST + c];
            h2[row * H2_ST + c] = (t - mean) * rs * g2[c] + be2[c];
        }
    }
    __syncthreads();

    // ---------------- Stage D: max over 32 points, final LayerNorm, store
    {
        const int col = tid;                      // 0..255
        float m = -3.402823466e+38f;
        #pragma unroll
        for (int r = 0; r < PATCH_SIZE; ++r)
            m = fmaxf(m, h2[r * H2_ST + col]);

        float sum = m, sq = m * m;
        #pragma unroll
        for (int msk = 1; msk < 32; msk <<= 1) {
            sum += __shfl_xor(sum, msk, 32);
            sq  += __shfl_xor(sq,  msk, 32);
        }
        if (lane == 0) { red[wave] = sum; red[8 + wave] = sq; }
        __syncthreads();
        float ts = 0.f, tq = 0.f;
        #pragma unroll
        for (int i = 0; i < 8; ++i) { ts += red[i]; tq += red[8 + i]; }
        const float mean = ts * (1.f / EMBED_DIM);
        const float var  = tq * (1.f / EMBED_DIM) - mean * mean;
        const float rs   = rsqrtf(var + LN_EPS);
        out[(size_t)blk * EMBED_DIM + col] = (m - mean) * rs * gn[col] + bn[col];
    }
}

// ---------------------------------------------------------------------------
extern "C" void kernel_launch(void* const* d_in, const int* in_sizes, int n_in,
                              void* d_out, int out_size, void* d_ws, size_t ws_size,
                              hipStream_t stream) {
    const float* x   = (const float*)d_in[0];
    const float* w1  = (const float*)d_in[1];
    const float* b1  = (const float*)d_in[2];
    const float* g1  = (const float*)d_in[3];
    const float* be1 = (const float*)d_in[4];
    const float* w2  = (const float*)d_in[5];
    const float* b2  = (const float*)d_in[6];
    const float* g2  = (const float*)d_in[7];
    const float* be2 = (const float*)d_in[8];
    const float* gn  = (const float*)d_in[9];
    const float* bn  = (const float*)d_in[10];
    float* out = (float*)d_out;

    _Float16* w2h = (_Float16*)d_ws;  // needs 128*256*2 = 64 KB of scratch

    repack_w2_kernel<<<(HIDDEN * EMBED_DIM) / 256, 256, 0, stream>>>(w2, w2h);
    patch_embed_kernel<<<BATCH * N_PATCHES, 256, 0, stream>>>(
        x, w1, b1, g1, be1, w2h, b2, g2, be2, gn, bn, out);
}
